// RNNBlock_29068338659427
// MI455X (gfx1250) — matrix operand order repaired and checked by reading, not compiled, not run on hardware
//
#include <hip/hip_runtime.h>

// Problem constants
#define T_   2048
#define B_   32
#define I_   512
#define H_   512
#define G_   1536          // 3*H
#define NCAT 2048          // 3*H + H  (gi columns + dense columns)
#define MTOT (T_*B_)       // 65536 rows of X

typedef __bf16 bf16;
typedef __attribute__((ext_vector_type(16))) __bf16       v16bf;
typedef __attribute__((ext_vector_type(8)))  float        v8f;
typedef __attribute__((ext_vector_type(4)))  unsigned int u32x4;
typedef __attribute__((ext_vector_type(4)))  float        f32x4;
typedef __attribute__((ext_vector_type(4)))  unsigned int v4u;
typedef __attribute__((ext_vector_type(8)))  int          v8i;
typedef __attribute__((ext_vector_type(4)))  int          v4i;

union FragAB { v16bf v; u32x4 u[2]; };  // 32 bytes: one WMMA A or B fragment per lane

// Workspace layout (bytes)
#define XB_OFF   ((size_t)0)                                  // bf16 X        [MTOT][512]   64 MB
#define WCAT_OFF (XB_OFF  + (size_t)MTOT*I_*2)                // bf16 Wcat     [2048][512]    2 MB
#define WHH_OFF  (WCAT_OFF + (size_t)NCAT*I_*2)               // bf16 W_hh     [1536][512]  1.5 MB
#define GI_OFF   (WHH_OFF + (size_t)G_*I_*2)                  // bf16 gi_all   [MTOT][1536] 201 MB
#define H0_OFF   (GI_OFF  + (size_t)MTOT*G_*2)                // f32 h ping    [32][512]
#define H1_OFF   (H0_OFF  + (size_t)B_*H_*4)                  // f32 h pong    [32][512]
#define BAR_OFF  (H1_OFF  + (size_t)B_*H_*4)                  // u32 barrier counter

// Scan-kernel LDS partition (dynamic shared memory)
#define SM_WHH   (96*512*2)            // 98304 B : this WG's 96 W_hh rows (r/z/n x 32 cols)
#define SM_H     (32*512*4)            // 65536 B : full h state, fp32
#define SM_GH    (32*96*4)             // 12288 B : gh tile, fp32
#define SMEM_BYTES (SM_WHH + SM_H + SM_GH)   // 176128 B (< 320 KB WGP LDS)

//--------------------------------------------------------------------------
// K1: convert inputs to bf16 layouts, init h buffers and barrier
//--------------------------------------------------------------------------
__global__ __launch_bounds__(256) void k_convert(
    const float* __restrict__ X, const float* __restrict__ Wih,
    const float* __restrict__ Whh, const float* __restrict__ Wdense,
    const float* __restrict__ h0, char* __restrict__ ws)
{
  bf16*  Xb    = (bf16*)(ws + XB_OFF);
  bf16*  Wcat  = (bf16*)(ws + WCAT_OFF);
  bf16*  Whhb  = (bf16*)(ws + WHH_OFF);
  float* hA    = (float*)(ws + H0_OFF);
  float* hB    = (float*)(ws + H1_OFF);
  unsigned* bar = (unsigned*)(ws + BAR_OFF);

  size_t tid    = (size_t)blockIdx.x * blockDim.x + threadIdx.x;
  size_t stride = (size_t)gridDim.x * blockDim.x;

  for (size_t i = tid; i < (size_t)MTOT * I_; i += stride) Xb[i] = (bf16)X[i];

  for (size_t i = tid; i < (size_t)NCAT * I_; i += stride) {
    size_t n = i >> 9, k = i & 511;
    float w = (n < (size_t)G_) ? Wih[i] : Wdense[(n - G_) * I_ + k];
    Wcat[i] = (bf16)w;
  }

  for (size_t i = tid; i < (size_t)G_ * I_; i += stride) Whhb[i] = (bf16)Whh[i];

  for (size_t i = tid; i < (size_t)B_ * H_; i += stride) { hA[i] = h0[i]; hB[i] = 0.f; }

  if (tid == 0) *bar = 0u;
}

//--------------------------------------------------------------------------
// K2: big parallel GEMM  [MTOT x 512] @ [512 x 2048]
//     cols [0,1536)  -> gi_all (bf16, +b_ih)   into ws
//     cols [1536,2048) -> dense (+b_dense)     into d_out
// Block: 256 threads (8 waves), tile 128(M) x 64(N); wave tile 16 x 64.
//--------------------------------------------------------------------------
__global__ __launch_bounds__(256) void k_gemm(
    char* __restrict__ ws, const float* __restrict__ b_ih,
    const float* __restrict__ b_dense, float* __restrict__ out)
{
  const bf16* Xb   = (const bf16*)(ws + XB_OFF);
  const bf16* Wcat = (const bf16*)(ws + WCAT_OFF);
  bf16*       gi   = (bf16*)(ws + GI_OFF);

  const int wave = threadIdx.x >> 5;
  const int lane = threadIdx.x & 31;
  const int half = lane >> 4;      // 0: lanes 0-15, 1: lanes 16-31
  const int r    = lane & 15;

  const int mTile = blockIdx.x * 128 + wave * 16;   // 16 rows per wave
  const int nBase = blockIdx.y * 64;                // 4 N-tiles per wave

  const v8f vz = {0.f,0.f,0.f,0.f,0.f,0.f,0.f,0.f};
  v8f acc[4] = { vz, vz, vz, vz };

  const bf16* arow = Xb + (size_t)(mTile + r) * I_;

  for (int k = 0; k < I_; k += 32) {
    // A fragment: lane holds M = mTile+r, K = {k0..k0+15}
    FragAB a;
    const int k0 = k + half * 16;
    a.u[0] = *(const u32x4*)(arow + k0);
    a.u[1] = *(const u32x4*)(arow + k0 + 8);
#pragma unroll
    for (int j = 0; j < 4; ++j) {
      // B fragment: lane holds column N = n (= W row n), 16 contiguous K values
      const int n = nBase + j * 16 + r;
      const bf16* brow = Wcat + (size_t)n * I_ + k + half * 16;
      FragAB b;
      b.u[0] = ((const u32x4*)brow)[0];
      b.u[1] = ((const u32x4*)brow)[1];
      acc[j] = __builtin_amdgcn_wmma_f32_16x16x32_bf16(
          false, a.v, false, b.v, (short)0, acc[j], false, false);
    }
  }

  // Epilogue: C layout -> lane col = r, VGPR v -> row v + 8*half
#pragma unroll
  for (int j = 0; j < 4; ++j) {
    const int n = nBase + j * 16 + r;
#pragma unroll
    for (int v = 0; v < 8; ++v) {
      const int m = mTile + v + 8 * half;
      const float val = acc[j][v];
      if (n < G_) {
        gi[(size_t)m * G_ + n] = (bf16)(val + b_ih[n]);
      } else {
        const int hcol = n - G_;
        out[(size_t)m * H_ + hcol] = val + b_dense[hcol];
      }
    }
  }
}

//--------------------------------------------------------------------------
// K3: persistent GRU scan. 16 WGs x 384 threads (12 waves).
// WG wg owns hidden cols [wg*32, wg*32+32): keeps its 96 W_hh rows in LDS
// (filled once via the Tensor Data Mover). Per step: stage h (LDS) ->
// WMMA gh tile (LDS) -> fp32 gates -> h ping-pong + out accumulate ->
// grid barrier.
//--------------------------------------------------------------------------
__global__ __launch_bounds__(384) void k_scan(
    char* __restrict__ ws, const float* __restrict__ b_hh, float* __restrict__ out)
{
  extern __shared__ char smem[];
  bf16*  WhhL = (bf16*)smem;                       // [96][512] at LDS offset 0
  float* hL   = (float*)(smem + SM_WHH);           // [32][512]
  float* ghL  = (float*)(smem + SM_WHH + SM_H);    // [32][96]

  const bf16* Whhb = (const bf16*)(ws + WHH_OFF);
  const bf16* gi   = (const bf16*)(ws + GI_OFF);
  float* hbuf0 = (float*)(ws + H0_OFF);
  float* hbuf1 = (float*)(ws + H1_OFF);
  unsigned* bar = (unsigned*)(ws + BAR_OFF);

  const int tid = threadIdx.x;
  const int wg  = blockIdx.x;        // 0..15
  const int colBase = wg * 32;

  // ---- Fill W_hh slice into LDS ----
  // Three 32x512 bf16 slabs (r/z/n gate rows for our columns), each row-major
  // contiguous: ideal Tensor Data Mover tiles.
#if __has_builtin(__builtin_amdgcn_tensor_load_to_lds)
  if (tid == 0) {
    for (int sec = 0; sec < 3; ++sec) {
      const unsigned long long ga = (unsigned long long)(ws + WHH_OFF)
          + ((unsigned long long)(sec * H_ + colBase) * I_) * 2ull;
      // D# group0: count=1 | lds_addr | global_addr[56:0] | type=2
      const v4u g0 = { 1u,
                       (unsigned)(sec * 32 * 512 * 2),
                       (unsigned)(ga & 0xFFFFFFFFull),
                       (unsigned)((ga >> 32) & 0x1FFFFFFull) | 0x80000000u };
      // D# group1: data_size=2B; tensor 512x32; tile 512x32; dim0_stride=512
      const v8i g1 = { (int)0x00010000,      // wg_mask=0, data_size=1 (2B)
                       (int)(512u << 16),    // tensor_dim0[15:0] in [63:48]
                       (int)(32u  << 16),    // tensor_dim1[15:0] in [95:80]
                       (int)(512u << 16),    // tile_dim0 in [127:112]
                       (int)32,              // tile_dim1 in [143:128]
                       (int)512,             // tensor_dim0_stride[31:0]
                       0, 0 };
      const v4i gz = { 0, 0, 0, 0 };
#if __clang_major__ >= 23
      const v8i gz8 = { 0,0,0,0,0,0,0,0 };
      __builtin_amdgcn_tensor_load_to_lds(g0, g1, gz, gz, gz8, 0);
#else
      __builtin_amdgcn_tensor_load_to_lds(g0, g1, gz, gz, 0);
#endif
    }
#if __has_builtin(__builtin_amdgcn_s_wait_tensorcnt)
    __builtin_amdgcn_s_wait_tensorcnt(0);
#endif
  }
#else
  for (int i = tid; i < 96 * 64; i += 384) {      // 64 u32x4 per 512-col row
    const int rowL = i >> 6, c = i & 63;
    const int gRow = (rowL >> 5) * H_ + colBase + (rowL & 31);
    ((u32x4*)(WhhL + rowL * 512))[c] = ((const u32x4*)(Whhb + (size_t)gRow * I_))[c];
  }
#endif
  __syncthreads();

  const int wave = tid >> 5;
  const int lane = tid & 31;
  const int half = lane >> 4;
  const int r    = lane & 15;
  const int mTile = wave & 1;        // 2 M-tiles (B=32)
  const int nTile = wave >> 1;       // 6 N-tiles (96 gh cols)

  const v8f vz = {0.f,0.f,0.f,0.f,0.f,0.f,0.f,0.f};

  for (int t = 0; t < T_; ++t) {
    const float* hc = (t & 1) ? hbuf1 : hbuf0;
    float*       hn = (t & 1) ? hbuf0 : hbuf1;

    // Stage full h into LDS (64 KB through L2)
    for (int i = tid; i < (B_ * H_) / 4; i += 384)
      ((f32x4*)hL)[i] = ((const f32x4*)hc)[i];

    // Prefetch next step's gi rows for our columns
    if (t + 1 < T_ && tid < 32) {
      const bf16* p = gi + ((size_t)(t + 1) * B_ + tid) * G_ + colBase;
      __builtin_prefetch((const void*)p, 0, 1);
      __builtin_prefetch((const void*)(p + 512), 0, 1);
      __builtin_prefetch((const void*)(p + 1024), 0, 1);
    }
    __syncthreads();

    // gh tile: [32 x 512] @ [512 x 96] -> wave computes one 16x16 C tile
    v8f acc = vz;
    const float* hrow  = hL + (mTile * 16 + r) * H_;
    const bf16*  bbase = WhhL + (nTile * 16 + r) * 512;
    for (int k = 0; k < H_; k += 32) {
      FragAB a;
      const int k0 = k + half * 16;
#pragma unroll
      for (int e = 0; e < 8; ++e) {
        a.v[e]     = (bf16)hrow[k0 + e];
        a.v[8 + e] = (bf16)hrow[k0 + 8 + e];
      }
      FragAB b;
      const u32x4* bp = (const u32x4*)(bbase + k + half * 16);
      b.u[0] = bp[0];
      b.u[1] = bp[1];
      acc = __builtin_amdgcn_wmma_f32_16x16x32_bf16(
          false, a.v, false, b.v, (short)0, acc, false, false);
    }
#pragma unroll
    for (int v = 0; v < 8; ++v)
      ghL[(mTile * 16 + v + 8 * half) * 96 + nTile * 16 + r] = acc[v];
    __syncthreads();

    // Gates (fp32) over 32x32 elements
    for (int i = tid; i < B_ * 32; i += 384) {
      const int bb = i >> 5, j = i & 31;
      const int col = colBase + j;
      const size_t girow = ((size_t)t * B_ + bb) * G_;
      const float gir = (float)gi[girow + col];
      const float giz = (float)gi[girow + 512 + col];
      const float gin = (float)gi[girow + 1024 + col];
      const float ghr = ghL[bb * 96 + j]      + b_hh[col];
      const float ghz = ghL[bb * 96 + 32 + j] + b_hh[512 + col];
      const float ghn = ghL[bb * 96 + 64 + j] + b_hh[1024 + col];
      const float rg = 1.f / (1.f + __expf(-(gir + ghr)));
      const float zg = 1.f / (1.f + __expf(-(giz + ghz)));
      const float ng = tanhf(gin + rg * ghn);
      const float hold = hL[bb * H_ + col];
      const float hv = (1.f - zg) * ng + zg * hold;
      hn[bb * H_ + col] = hv;
      const size_t oidx = ((size_t)t * B_ + bb) * H_ + col;
      out[oidx] = out[oidx] + hv;                       // + dense from K2
      if (t == T_ - 1) out[(size_t)T_ * B_ * H_ + bb * H_ + col] = hv;  // h_last
    }

    // Grid barrier across 16 WGs (monotonic counter, 16*(t+1) target)
    __syncthreads();
    __threadfence();
    if (tid == 0) {
      atomicAdd(bar, 1u);
      const unsigned target = 16u * (unsigned)(t + 1);
      volatile unsigned* vb = bar;
      while (*vb < target) __builtin_amdgcn_s_sleep(1);
    }
    __syncthreads();
    __threadfence();
  }
}

//--------------------------------------------------------------------------
extern "C" void kernel_launch(void* const* d_in, const int* in_sizes, int n_in,
                              void* d_out, int out_size, void* d_ws, size_t ws_size,
                              hipStream_t stream)
{
  (void)in_sizes; (void)n_in; (void)out_size; (void)ws_size;
  const float* X       = (const float*)d_in[0];
  const float* h0      = (const float*)d_in[1];
  const float* W_ih    = (const float*)d_in[2];
  const float* W_hh    = (const float*)d_in[3];
  const float* b_ih    = (const float*)d_in[4];
  const float* b_hh    = (const float*)d_in[5];
  const float* W_dense = (const float*)d_in[6];
  const float* b_dense = (const float*)d_in[7];
  float* out = (float*)d_out;
  char*  ws  = (char*)d_ws;

  hipFuncSetAttribute((const void*)k_scan,
                      hipFuncAttributeMaxDynamicSharedMemorySize, SMEM_BYTES);

  k_convert<<<4096, 256, 0, stream>>>(X, W_ih, W_hh, W_dense, h0, ws);

  dim3 g2(MTOT / 128, NCAT / 64);   // 512 x 32 blocks
  k_gemm<<<g2, 256, 0, stream>>>(ws, b_ih, b_dense, out);

  k_scan<<<16, 384, SMEM_BYTES, stream>>>(ws, b_hh, out);
}
